// HierarchicalCrossAttentionPrior_54520314855958
// MI455X (gfx1250) — compile-verified, matrix operand order checked by reading, not run on hardware
//
#include <hip/hip_runtime.h>
#include <math.h>

typedef __bf16 bf16;
typedef __attribute__((ext_vector_type(16))) __bf16 v16bf;
typedef __attribute__((ext_vector_type(8)))  __bf16 v8bf;
typedef __attribute__((ext_vector_type(8)))  float  v8f;

static_assert(sizeof(bf16) == 2, "bf16 size");

union FragAB { v16bf v; v8bf h[2]; };

__device__ __forceinline__ v8f wmma_bf16(v16bf a, v16bf b, v8f c) {
  return __builtin_amdgcn_wmma_f32_16x16x32_bf16(false, a, false, b, (short)0, c, false, false);
}

__device__ __forceinline__ v8f zero8() {
  v8f z;
  #pragma unroll
  for (int i = 0; i < 8; i++) z[i] = 0.f;
  return z;
}

__device__ __forceinline__ float wave_sum(float s) {
  #pragma unroll
  for (int o = 16; o > 0; o >>= 1) s += __shfl_xor(s, o, 32);
  return s;
}

// Async global->LDS 16B copy (CDNA5 GLOBAL_LOAD_ASYNC_TO_LDS_B128, ASYNCcnt).
// VDST operand = per-lane LDS byte offset; low 32 bits of a generic LDS pointer
// are the wave-logical LDS offset (flat aperture only occupies addr[63:32]).
__device__ __forceinline__ void async_copy16(void* lds_dst, const void* gsrc) {
  const unsigned ldsoff = (unsigned)(size_t)lds_dst;
  const unsigned long long ga = (unsigned long long)(size_t)gsrc;
  asm volatile("global_load_async_to_lds_b128 %0, %1, off"
               :: "v"(ldsoff), "v"(ga) : "memory");
}
__device__ __forceinline__ void async_wait0() {
  asm volatile("s_wait_asynccnt 0x0" ::: "memory");
}

// ---------------------------------------------------------------------------
// Generic GEMM: C[M,N] = act(A[M,K](bf16) @ Wt[N,K]^T(bf16) + bias[N])
// Tile 128x128, 256 threads = 8 waves (4 M-waves x 2 N-waves), wave = 32x64.
// Double-buffered K tiles filled with async global->LDS copies.
// ACT: 0=none, 1=relu, 2=gelu(exact). OUTBF: bf16 out else f32 out.
// ---------------------------------------------------------------------------
#define LDT 40   // K-tile 32 + 8 pad (80B rows: 16B aligned, conflict-free)

template <int ACT, int OUTBF>
__global__ __launch_bounds__(256) void gemm_bf16(
    const bf16* __restrict__ A, const bf16* __restrict__ Wt,
    const float* __restrict__ bias, float* __restrict__ outF,
    bf16* __restrict__ outB, int M, int N, int K)
{
  __shared__ __align__(16) bf16 Ash[2][128 * LDT];
  __shared__ __align__(16) bf16 Bsh[2][128 * LDT];
  const int tid = threadIdx.x, lane = tid & 31, w = tid >> 5;
  const int mw = w & 3, nw = w >> 2;
  const int m0 = blockIdx.x * 128, n0 = blockIdx.y * 128;
  const int rsel = lane & 15, kh = (lane >> 4) * 8;

  v8f acc[2][4];
  #pragma unroll
  for (int i = 0; i < 2; i++)
    #pragma unroll
    for (int j = 0; j < 4; j++) acc[i][j] = zero8();

  const int lr = tid >> 1, lc = (tid & 1) * 16;
  const bf16* gA = A  + (size_t)(m0 + lr) * K + lc;
  const bf16* gB = Wt + (size_t)(n0 + lr) * K + lc;

  // prologue: fill buffer 0
  async_copy16(&Ash[0][lr * LDT + lc],     gA);
  async_copy16(&Ash[0][lr * LDT + lc + 8], gA + 8);
  async_copy16(&Bsh[0][lr * LDT + lc],     gB);
  async_copy16(&Bsh[0][lr * LDT + lc + 8], gB + 8);
  async_wait0();
  __syncthreads();

  int buf = 0;
  for (int k0 = 0; k0 < K; k0 += 32) {
    if (k0 + 32 < K) {  // prefetch next K tile into the other buffer
      const int nb = buf ^ 1;
      async_copy16(&Ash[nb][lr * LDT + lc],     gA + k0 + 32);
      async_copy16(&Ash[nb][lr * LDT + lc + 8], gA + k0 + 40);
      async_copy16(&Bsh[nb][lr * LDT + lc],     gB + k0 + 32);
      async_copy16(&Bsh[nb][lr * LDT + lc + 8], gB + k0 + 40);
    }

    FragAB a0, a1;
    const bf16* ap0 = &Ash[buf][(mw * 16 + rsel) * LDT + kh];
    a0.h[0] = *(const v8bf*)ap0; a0.h[1] = *(const v8bf*)(ap0 + 16);
    const bf16* ap1 = &Ash[buf][(64 + mw * 16 + rsel) * LDT + kh];
    a1.h[0] = *(const v8bf*)ap1; a1.h[1] = *(const v8bf*)(ap1 + 16);
    #pragma unroll
    for (int j = 0; j < 4; j++) {
      FragAB b;
      const bf16* bp = &Bsh[buf][(nw * 64 + j * 16 + rsel) * LDT + kh];
      b.h[0] = *(const v8bf*)bp; b.h[1] = *(const v8bf*)(bp + 16);
      acc[0][j] = wmma_bf16(a0.v, b.v, acc[0][j]);
      acc[1][j] = wmma_bf16(a1.v, b.v, acc[1][j]);
    }

    async_wait0();     // own prefetch landed
    __syncthreads();   // everyone's landed; all reads of old buffer done
    buf ^= 1;
  }

  const int col = lane & 15, rbase = 8 * (lane >> 4);
  #pragma unroll
  for (int i = 0; i < 2; i++) {
    const int rm = m0 + i * 64 + mw * 16 + rbase;
    #pragma unroll
    for (int j = 0; j < 4; j++) {
      const int cn = n0 + nw * 64 + j * 16 + col;
      const float bv = bias[cn];
      #pragma unroll
      for (int e = 0; e < 8; e++) {
        float vv = acc[i][j][e] + bv;
        if (ACT == 1) vv = fmaxf(vv, 0.f);
        else if (ACT == 2) vv = 0.5f * vv * (1.f + erff(vv * 0.70710678118f));
        const size_t idx = (size_t)(rm + e) * N + cn;
        if (OUTBF) outB[idx] = (bf16)vv;
        else       outF[idx] = vv;
      }
    }
  }
}

// ---------------------------------------------------------------------------
// Fused FFN: out[M,128] = relu(A[M,128] @ W1t[2048,128]^T + b1) @ W2t[128,2048]^T + b2
// Hidden (M x 2048) never touches HBM: processed in 128-wide chunks via LDS.
// ---------------------------------------------------------------------------
#define LDF 136  // 128 + 8 pad (272B rows: 16B aligned, conflict-free)

__global__ __launch_bounds__(256) void ff_fused(
    const bf16* __restrict__ A, const bf16* __restrict__ W1t,
    const float* __restrict__ b1, const bf16* __restrict__ W2t,
    const float* __restrict__ b2, float* __restrict__ out)
{
  __shared__ __align__(16) bf16 Wbuf[128 * LDF];
  __shared__ __align__(16) bf16 Hs[64 * LDF];
  const int tid = threadIdx.x, lane = tid & 31, w = tid >> 5;
  const int mw = w & 3, nw = w >> 2;
  const int m0 = blockIdx.x * 64;
  const int rsel = lane & 15, kh = (lane >> 4) * 8;
  const int col = lane & 15, rbase = 8 * (lane >> 4);

  // Per-wave A rows live in registers for the whole kernel (K = 128 = 4 frags)
  FragAB afr[4];
  {
    const bf16* ar = A + (size_t)(m0 + mw * 16 + rsel) * 128;
    #pragma unroll
    for (int ks = 0; ks < 4; ks++) {
      const bf16* p = ar + ks * 32 + kh;
      afr[ks].h[0] = *(const v8bf*)p; afr[ks].h[1] = *(const v8bf*)(p + 16);
    }
  }
  v8f acc2[4];
  #pragma unroll
  for (int j = 0; j < 4; j++) acc2[j] = zero8();

  const int lr = tid >> 1, lc = (tid & 1) * 64;
  for (int ch = 0; ch < 16; ch++) {
    {  // async load W1 chunk (rows ch*128..+127 of W1t, K=128)
      const bf16* g1 = W1t + (size_t)(ch * 128 + lr) * 128 + lc;
      #pragma unroll
      for (int u = 0; u < 8; u++)
        async_copy16(&Wbuf[lr * LDF + lc + u * 8], g1 + u * 8);
    }
    async_wait0();
    __syncthreads();

    v8f acc1[4];
    #pragma unroll
    for (int j = 0; j < 4; j++) acc1[j] = zero8();
    #pragma unroll
    for (int ks = 0; ks < 4; ks++) {
      #pragma unroll
      for (int j = 0; j < 4; j++) {
        FragAB b;
        const bf16* bp = &Wbuf[(nw * 64 + j * 16 + rsel) * LDF + ks * 32 + kh];
        b.h[0] = *(const v8bf*)bp; b.h[1] = *(const v8bf*)(bp + 16);
        acc1[j] = wmma_bf16(afr[ks].v, b.v, acc1[j]);
      }
    }
    __syncthreads();  // all waves done reading Wbuf

    // bias + relu -> H (LDS, bf16); refill Wbuf with W2 chunk (async)
    #pragma unroll
    for (int j = 0; j < 4; j++) {
      const int cn = nw * 64 + j * 16 + col;
      const float bb = b1[ch * 128 + cn];
      #pragma unroll
      for (int e = 0; e < 8; e++) {
        float vv = fmaxf(acc1[j][e] + bb, 0.f);
        Hs[(mw * 16 + rbase + e) * LDF + cn] = (bf16)vv;
      }
    }
    {
      const bf16* g2 = W2t + (size_t)lr * 2048 + ch * 128 + lc;
      #pragma unroll
      for (int u = 0; u < 8; u++)
        async_copy16(&Wbuf[lr * LDF + lc + u * 8], g2 + u * 8);
    }
    async_wait0();
    __syncthreads();

    #pragma unroll
    for (int ks = 0; ks < 4; ks++) {
      FragAB ha;
      const bf16* hp = &Hs[(mw * 16 + rsel) * LDF + ks * 32 + kh];
      ha.h[0] = *(const v8bf*)hp; ha.h[1] = *(const v8bf*)(hp + 16);
      #pragma unroll
      for (int j = 0; j < 4; j++) {
        FragAB b;
        const bf16* bp = &Wbuf[(nw * 64 + j * 16 + rsel) * LDF + ks * 32 + kh];
        b.h[0] = *(const v8bf*)bp; b.h[1] = *(const v8bf*)(bp + 16);
        acc2[j] = wmma_bf16(ha.v, b.v, acc2[j]);
      }
    }
    __syncthreads();
  }

  #pragma unroll
  for (int j = 0; j < 4; j++) {
    const int cn = nw * 64 + j * 16 + col;
    const float bb = b2[cn];
    #pragma unroll
    for (int e = 0; e < 8; e++)
      out[(size_t)(m0 + mw * 16 + rbase + e) * 128 + cn] = acc2[j][e] + bb;
  }
}

// ---------------------------------------------------------------------------
// Weight repack: f32 (b, K, N) row-major  ->  bf16 (b, N, K)
// ---------------------------------------------------------------------------
__global__ void k_repack(const float* __restrict__ in, bf16* __restrict__ out,
                         int K, int N)
{
  const size_t i = (size_t)blockIdx.x * 256 + threadIdx.x;
  const size_t per = (size_t)K * N;
  const size_t b = i / per, r = i - b * per;
  const int n = (int)(r / K), k = (int)(r - (size_t)n * K);
  out[i] = (bf16)in[b * per + (size_t)k * N + n];
}

// gp_w (G=8, D=1024, GD=128) -> Wt[n=g*128+k][d]
__global__ void k_repack_gp(const float* __restrict__ gp, bf16* __restrict__ out)
{
  const size_t i = (size_t)blockIdx.x * 256 + threadIdx.x;  // over 1024*1024
  const int n = (int)(i >> 10), d = (int)(i & 1023);
  const int g = n >> 7, k = n & 127;
  out[i] = (bf16)gp[((size_t)g * 1024 + d) * 128 + k];
}

__global__ void k_cast(const float* __restrict__ in, bf16* __restrict__ out)
{
  const size_t i = (size_t)blockIdx.x * 256 + threadIdx.x;
  out[i] = (bf16)in[i];
}

// Sinusoidal time embedding: [sin(t*f_0..511), cos(t*f_0..511)]
__global__ void k_time_embed(const float* __restrict__ t, bf16* __restrict__ te)
{
  const size_t i = (size_t)blockIdx.x * 256 + threadIdx.x;
  const int b = (int)(i >> 10), d = (int)(i & 1023);
  const int h = d & 511;
  const float fr = expf((float)h * (-9.210340371976184f / 511.0f));
  const float a = t[b] * fr;
  te[i] = (bf16)((d < 512) ? sinf(a) : cosf(a));
}

// cond = user_table[uid] + score_table[score] + te_mlp_out
__global__ void k_cond(const float* __restrict__ te, const float* __restrict__ ut,
                       const float* __restrict__ st, const int* __restrict__ uid,
                       const int* __restrict__ sc, bf16* __restrict__ cond)
{
  const size_t i = (size_t)blockIdx.x * 256 + threadIdx.x;
  const int b = (int)(i >> 10), d = (int)(i & 1023);
  cond[i] = (bf16)(te[i] + ut[(size_t)uid[b] * 1024 + d] + st[(size_t)sc[b] * 1024 + d]);
}

// xg = LN(xg + ca; g1,b1);  q = LN(xg; g2,b2) -> bf16.  One wave per 128-row.
__global__ __launch_bounds__(256) void k_ln_gca(
    const float* xg_in, const float* __restrict__ ca,
    const float* __restrict__ g1, const float* __restrict__ b1,
    const float* __restrict__ g2, const float* __restrict__ b2,
    float* xg_out, bf16* __restrict__ q_out)
{
  const int lane = threadIdx.x & 31;
  const size_t base = (((size_t)blockIdx.x * 256 + threadIdx.x) >> 5) * 128;
  float v[4]; float s = 0.f;
  #pragma unroll
  for (int u = 0; u < 4; u++) { const int c = lane + 32 * u; v[u] = xg_in[base + c] + ca[base + c]; s += v[u]; }
  const float mean = wave_sum(s) * (1.f / 128.f);
  float s2 = 0.f;
  #pragma unroll
  for (int u = 0; u < 4; u++) { const float d = v[u] - mean; s2 += d * d; }
  const float inv = rsqrtf(wave_sum(s2) * (1.f / 128.f) + 1e-5f);
  float y[4]; float sy = 0.f;
  #pragma unroll
  for (int u = 0; u < 4; u++) {
    const int c = lane + 32 * u;
    y[u] = (v[u] - mean) * inv * g1[c] + b1[c];
    xg_out[base + c] = y[u]; sy += y[u];
  }
  const float m2 = wave_sum(sy) * (1.f / 128.f);
  float s3 = 0.f;
  #pragma unroll
  for (int u = 0; u < 4; u++) { const float d = y[u] - m2; s3 += d * d; }
  const float inv2 = rsqrtf(wave_sum(s3) * (1.f / 128.f) + 1e-5f);
  #pragma unroll
  for (int u = 0; u < 4; u++) {
    const int c = lane + 32 * u;
    q_out[base + c] = (bf16)((y[u] - m2) * inv2 * g2[c] + b2[c]);
  }
}

// h = xg + attn; r = relu(LN(h; g,b)) -> bf16
__global__ __launch_bounds__(256) void k_ln_enc2(
    const float* __restrict__ xg, const float* __restrict__ attn,
    const float* __restrict__ g, const float* __restrict__ b,
    float* __restrict__ h_out, bf16* __restrict__ r_out)
{
  const int lane = threadIdx.x & 31;
  const size_t base = (((size_t)blockIdx.x * 256 + threadIdx.x) >> 5) * 128;
  float v[4]; float s = 0.f;
  #pragma unroll
  for (int u = 0; u < 4; u++) { const int c = lane + 32 * u; v[u] = xg[base + c] + attn[base + c]; h_out[base + c] = v[u]; s += v[u]; }
  const float mean = wave_sum(s) * (1.f / 128.f);
  float s2 = 0.f;
  #pragma unroll
  for (int u = 0; u < 4; u++) { const float d = v[u] - mean; s2 += d * d; }
  const float inv = rsqrtf(wave_sum(s2) * (1.f / 128.f) + 1e-5f);
  #pragma unroll
  for (int u = 0; u < 4; u++) {
    const int c = lane + 32 * u;
    r_out[base + c] = (bf16)fmaxf((v[u] - mean) * inv * g[c] + b[c], 0.f);
  }
}

// xg = LN(xg + h + ff; g,b)
__global__ __launch_bounds__(256) void k_ln_gsn(
    const float* xg_in, const float* __restrict__ h, const float* __restrict__ ff,
    const float* __restrict__ g, const float* __restrict__ b, float* xg_out)
{
  const int lane = threadIdx.x & 31;
  const size_t base = (((size_t)blockIdx.x * 256 + threadIdx.x) >> 5) * 128;
  float v[4]; float s = 0.f;
  #pragma unroll
  for (int u = 0; u < 4; u++) { const int c = lane + 32 * u; v[u] = xg_in[base + c] + h[base + c] + ff[base + c]; s += v[u]; }
  const float mean = wave_sum(s) * (1.f / 128.f);
  float s2 = 0.f;
  #pragma unroll
  for (int u = 0; u < 4; u++) { const float d = v[u] - mean; s2 += d * d; }
  const float inv = rsqrtf(wave_sum(s2) * (1.f / 128.f) + 1e-5f);
  #pragma unroll
  for (int u = 0; u < 4; u++) {
    const int c = lane + 32 * u;
    xg_out[base + c] = (v[u] - mean) * inv * g[c] + b[c];
  }
}

// 1024-wide LN: xgl = LN(xg + gc; gln) -> bf16. One 256-thread block per row.
__global__ __launch_bounds__(256) void k_ln_glob(
    const float* __restrict__ xg, const float* __restrict__ gc,
    const float* __restrict__ g, const float* __restrict__ b, bf16* __restrict__ out)
{
  __shared__ float red[8];
  __shared__ float sM, sI;
  const int tid = threadIdx.x, lane = tid & 31, w = tid >> 5;
  const size_t base = (size_t)blockIdx.x * 1024;
  float v[4]; float s = 0.f;
  #pragma unroll
  for (int u = 0; u < 4; u++) { const int c = tid + 256 * u; v[u] = xg[base + c] + gc[base + c]; s += v[u]; }
  s = wave_sum(s);
  if (lane == 0) red[w] = s;
  __syncthreads();
  if (tid == 0) { float tt = 0.f; for (int i = 0; i < 8; i++) tt += red[i]; sM = tt * (1.f / 1024.f); }
  __syncthreads();
  const float mean = sM;
  float s2 = 0.f;
  #pragma unroll
  for (int u = 0; u < 4; u++) { const float d = v[u] - mean; s2 += d * d; }
  s2 = wave_sum(s2);
  if (lane == 0) red[w] = s2;
  __syncthreads();
  if (tid == 0) { float tt = 0.f; for (int i = 0; i < 8; i++) tt += red[i]; sI = rsqrtf(tt * (1.f / 1024.f) + 1e-5f); }
  __syncthreads();
  const float inv = sI;
  #pragma unroll
  for (int u = 0; u < 4; u++) {
    const int c = tid + 256 * u;
    out[base + c] = (bf16)((v[u] - mean) * inv * g[c] + b[c]);
  }
}

// ---------------------------------------------------------------------------
extern "C" void kernel_launch(void* const* d_in, const int* in_sizes, int n_in,
                              void* d_out, int out_size, void* d_ws, size_t ws_size,
                              hipStream_t stream)
{
  (void)in_sizes; (void)n_in; (void)out_size; (void)ws_size;
  const float* x          = (const float*)d_in[0];
  const float* t          = (const float*)d_in[1];
  const int*   uid        = (const int*)d_in[2];
  const int*   sc         = (const int*)d_in[3];
  const float* user_table = (const float*)d_in[4];
  const float* score_table= (const float*)d_in[5];
  const float* tp_w1 = (const float*)d_in[6];  const float* tp_b1 = (const float*)d_in[7];
  const float* tp_w2 = (const float*)d_in[8];  const float* tp_b2 = (const float*)d_in[9];
  const float* gp_w  = (const float*)d_in[10]; const float* gp_b  = (const float*)d_in[11];
  // q/k weights (12,13,16,17,20,21,24,25,28,29,32,33) are dead: softmax over len-1 == 1
  const float* gca_wv = (const float*)d_in[14]; const float* gca_wo = (const float*)d_in[15];
  const float* gca_bv = (const float*)d_in[18]; const float* gca_bo = (const float*)d_in[19];
  const float* enc_wv = (const float*)d_in[22]; const float* enc_wo = (const float*)d_in[23];
  const float* enc_bv = (const float*)d_in[26]; const float* enc_bo = (const float*)d_in[27];
  const float* glob_wv= (const float*)d_in[30]; const float* glob_wo= (const float*)d_in[31];
  const float* glob_bv= (const float*)d_in[34]; const float* glob_bo= (const float*)d_in[35];
  const float* eln1_g = (const float*)d_in[36]; const float* eln1_b = (const float*)d_in[37];
  const float* eln2_g = (const float*)d_in[38]; const float* eln2_b = (const float*)d_in[39];
  const float* gcn_g  = (const float*)d_in[40]; const float* gcn_b  = (const float*)d_in[41];
  const float* gsn_g  = (const float*)d_in[42]; const float* gsn_b  = (const float*)d_in[43];
  const float* gln_g  = (const float*)d_in[44]; const float* gln_b  = (const float*)d_in[45];
  const float* ff_w1  = (const float*)d_in[46]; const float* ff_b1  = (const float*)d_in[47];
  const float* ff_w2  = (const float*)d_in[48]; const float* ff_b2  = (const float*)d_in[49];
  const float* op_w1  = (const float*)d_in[50]; const float* op_b1  = (const float*)d_in[51];
  const float* op_w2  = (const float*)d_in[52]; const float* op_b2  = (const float*)d_in[53];
  float* out = (float*)d_out;

  const size_t Bsz = 16384;
  char* wp = (char*)d_ws; size_t off = 0;
  auto alloc = [&](size_t bytes) -> void* {
    void* p = wp + off; off += (bytes + 255) & ~(size_t)255; return p;
  };
  bf16* w_tp1t = (bf16*)alloc(4096ull * 1024 * 2);
  bf16* w_tp2t = (bf16*)alloc(1024ull * 4096 * 2);
  bf16* w_gpt  = (bf16*)alloc(1024ull * 1024 * 2);
  bf16* w_gcav = (bf16*)alloc(6ull * 128 * 128 * 2);
  bf16* w_gcao = (bf16*)alloc(6ull * 128 * 128 * 2);
  bf16* w_encv = (bf16*)alloc(6ull * 128 * 128 * 2);
  bf16* w_enco = (bf16*)alloc(6ull * 128 * 128 * 2);
  bf16* w_ff1  = (bf16*)alloc(6ull * 2048 * 128 * 2);
  bf16* w_ff2  = (bf16*)alloc(6ull * 128 * 2048 * 2);
  bf16* w_glv  = (bf16*)alloc(6ull * 1024 * 1024 * 2);
  bf16* w_glo  = (bf16*)alloc(6ull * 1024 * 1024 * 2);
  bf16* w_op1  = (bf16*)alloc(2048ull * 1024 * 2);
  bf16* w_op2  = (bf16*)alloc(1024ull * 2048 * 2);
  bf16* t0b   = (bf16*)alloc(Bsz * 1024 * 2);
  bf16* qb    = (bf16*)alloc(Bsz * 1024 * 2);
  bf16* t3b   = (bf16*)alloc(Bsz * 1024 * 2);
  bf16* condb = (bf16*)alloc(Bsz * 1024 * 2);
  bf16* cgb   = (bf16*)alloc(Bsz * 1024 * 2);
  bf16* bigb  = (bf16*)alloc(Bsz * 4096 * 2);
  float* xg = (float*)alloc(Bsz * 1024 * 4);
  float* t1 = (float*)alloc(Bsz * 1024 * 4);
  float* t2 = (float*)alloc(Bsz * 1024 * 4);

  auto rp = [&](const float* in, bf16* o, long bcnt, int K, int N) {
    long tot = bcnt * (long)K * N;
    k_repack<<<(int)(tot / 256), 256, 0, stream>>>(in, o, K, N);
  };
  rp(tp_w1, w_tp1t, 1, 1024, 4096);
  rp(tp_w2, w_tp2t, 1, 4096, 1024);
  k_repack_gp<<<1024 * 1024 / 256, 256, 0, stream>>>(gp_w, w_gpt);
  rp(gca_wv, w_gcav, 6, 128, 128);   rp(gca_wo, w_gcao, 6, 128, 128);
  rp(enc_wv, w_encv, 6, 128, 128);   rp(enc_wo, w_enco, 6, 128, 128);
  rp(ff_w1,  w_ff1,  6, 128, 2048);  rp(ff_w2,  w_ff2,  6, 2048, 128);
  rp(glob_wv, w_glv, 6, 1024, 1024); rp(glob_wo, w_glo, 6, 1024, 1024);
  rp(op_w1, w_op1, 1, 1024, 2048);   rp(op_w2, w_op2, 1, 2048, 1024);

  const int NB = (int)(Bsz * 1024 / 256);

  // time-embedding MLP -> cond
  k_time_embed<<<NB, 256, 0, stream>>>(t, t0b);
  gemm_bf16<2, 1><<<dim3(128, 32), 256, 0, stream>>>(t0b, w_tp1t, tp_b1, nullptr, bigb, 16384, 4096, 1024);
  gemm_bf16<0, 0><<<dim3(128, 8),  256, 0, stream>>>(bigb, w_tp2t, tp_b2, t1, nullptr, 16384, 1024, 4096);
  k_cond<<<NB, 256, 0, stream>>>(t1, user_table, score_table, uid, sc, condb);

  // loop-invariant group projections
  gemm_bf16<0, 1><<<dim3(128, 8), 256, 0, stream>>>(condb, w_gpt, gp_b, nullptr, cgb, 16384, 1024, 1024);
  k_cast<<<NB, 256, 0, stream>>>(x, t3b);
  gemm_bf16<0, 0><<<dim3(128, 8), 256, 0, stream>>>(t3b, w_gpt, gp_b, xg, nullptr, 16384, 1024, 1024);

  for (int l = 0; l < 6; l++) {
    // per-group cross attention (len-1 softmax => v/o projections only, input = cg)
    gemm_bf16<0, 1><<<dim3(1024, 1), 256, 0, stream>>>(cgb, w_gcav + (size_t)l * 128 * 128, gca_bv + l * 128, nullptr, t0b, 131072, 128, 128);
    gemm_bf16<0, 0><<<dim3(1024, 1), 256, 0, stream>>>(t0b, w_gcao + (size_t)l * 128 * 128, gca_bo + l * 128, t1, nullptr, 131072, 128, 128);
    k_ln_gca<<<16384, 256, 0, stream>>>(xg, t1, gcn_g + l * 128, gcn_b + l * 128, eln1_g + l * 128, eln1_b + l * 128, xg, qb);
    // encoder self-attention (v/o only, input = q)
    gemm_bf16<0, 1><<<dim3(1024, 1), 256, 0, stream>>>(qb, w_encv + (size_t)l * 128 * 128, enc_bv + l * 128, nullptr, t0b, 131072, 128, 128);
    gemm_bf16<0, 0><<<dim3(1024, 1), 256, 0, stream>>>(t0b, w_enco + (size_t)l * 128 * 128, enc_bo + l * 128, t1, nullptr, 131072, 128, 128);
    k_ln_enc2<<<16384, 256, 0, stream>>>(xg, t1, eln2_g + l * 128, eln2_b + l * 128, t2, t3b);
    // fused 128 -> 2048 -> 128 FFN
    ff_fused<<<2048, 256, 0, stream>>>(t3b, w_ff1 + (size_t)l * 2048 * 128, ff_b1 + l * 2048, w_ff2 + (size_t)l * 128 * 2048, ff_b2 + l * 128, t1);
    k_ln_gsn<<<16384, 256, 0, stream>>>(xg, t2, t1, gsn_g + l * 128, gsn_b + l * 128, xg);
    // global cross attention (v/o only, input = cond)
    gemm_bf16<0, 1><<<dim3(128, 8), 256, 0, stream>>>(condb, w_glv + (size_t)l * 1024 * 1024, glob_bv + l * 1024, nullptr, t3b, 16384, 1024, 1024);
    gemm_bf16<0, 0><<<dim3(128, 8), 256, 0, stream>>>(t3b, w_glo + (size_t)l * 1024 * 1024, glob_bo + l * 1024, t1, nullptr, 16384, 1024, 1024);
    k_ln_glob<<<16384, 256, 0, stream>>>(xg, t1, gln_g + l * 1024, gln_b + l * 1024, t3b);
    gemm_bf16<0, 0><<<dim3(128, 8), 256, 0, stream>>>(t3b, w_gpt, gp_b, xg, nullptr, 16384, 1024, 1024);
  }

  // output MLP
  k_cast<<<NB, 256, 0, stream>>>(xg, t3b);
  gemm_bf16<2, 1><<<dim3(128, 16), 256, 0, stream>>>(t3b, w_op1, op_b1, nullptr, bigb, 16384, 2048, 1024);
  gemm_bf16<0, 0><<<dim3(128, 8),  256, 0, stream>>>(bigb, w_op2, op_b2, out, nullptr, 16384, 1024, 2048);
}
